// BatchProgramClassifier_84645215470011
// MI455X (gfx1250) — matrix operand order repaired and checked by reading, not compiled, not run on hardware
//
#include <hip/hip_runtime.h>
#include <hip/hip_bf16.h>

// ------------------------- model dimensions -------------------------
namespace mc {
constexpr int  E   = 128;
constexpr int  E3  = 384;
constexpr int  FF  = 2048;
constexpr int  Bn  = 128;
constexpr int  NS  = 128;
constexpr int  DEP = 8;
constexpr int  Mp  = 3;
constexpr int  SC  = 128;
constexpr int  TD  = 512;
constexpr int  Lc  = 104;
constexpr int  Hh  = 4;
constexpr int  DH  = 32;
constexpr long NSB = (long)Bn * NS;   // 16384 tree rows
constexpr long PB  = (long)Bn * Mp;   // 384 cfg sequences
constexpr long PT  = PB * SC;         // 49152 cfg tokens
}

typedef _Float16 half_t;
typedef __attribute__((ext_vector_type(16))) _Float16 v16h;
typedef __attribute__((ext_vector_type(8)))  _Float16 v8h;
typedef __attribute__((ext_vector_type(8)))  float    v8f;

#define DEV static __device__ __forceinline__

DEV v8f vzero8() {
  v8f v;
#pragma unroll
  for (int i = 0; i < 8; ++i) v[i] = 0.f;
  return v;
}

DEV float sigmoidf(float x) { return 1.f / (1.f + expf(-x)); }

// WMMA D = A x B + C wrapper (codegen-confirmed builtin)
DEV v8f wmma_f16(v16h a, v16h b, v8f c) {
  return __builtin_amdgcn_wmma_f32_16x16x32_f16(false, a, false, b, (short)0, c,
                                                false, false);
}

// Pack A operand from a per-lane base pointer (two 16B chunks, +0 and +8 halfs)
DEV v16h pack_a(const half_t* p) {
  v8h lo = *(const v8h*)p;
  v8h hi = *(const v8h*)(p + 8);
  v16h r;
#pragma unroll
  for (int i = 0; i < 8; ++i) { r[i] = lo[i]; r[8 + i] = hi[i]; }
  return r;
}

// Pack B operand from a per-lane base pointer (one contiguous 32B chunk)
DEV v16h pack_b(const half_t* p) {
  v8h lo = *(const v8h*)p;
  v8h hi = *(const v8h*)(p + 8);
  v16h r;
#pragma unroll
  for (int i = 0; i < 8; ++i) { r[i] = lo[i]; r[8 + i] = hi[i]; }
  return r;
}

// A operand: 16x32 f16 tile, row-major source, row = lane%16.
// Lane<16 holds K=[0,16); lane>=16 holds K=[16,32).
DEV v16h wmma_load_a(const half_t* A, long lda) {
  const int lane = threadIdx.x & 31;
  return pack_a(A + (long)(lane & 15) * lda + ((lane < 16) ? 0 : 16));
}

// B operand: W stored [N,K] row-major (i.e. weights [out,in] for x@W.T).
// Lane holds column n = n0 + lane%16, contiguous K chunk of 16 at k0 in {0,16}.
DEV v16h wmma_load_bu(const half_t* W, long ldw, int n0, int k) {
  const int lane = threadIdx.x & 31;
  return pack_b(W + (long)(n0 + (lane & 15)) * ldw + k + ((lane < 16) ? 0 : 16));
}

// C/D tile store: lane holds column n=lane%16, rows v+(lane>=16?8:0)
DEV void gemm_store(v8f acc, float* C32, half_t* C16, const float* bias,
                    long m0, int n0, int N, int act) {
  const int lane = threadIdx.x & 31;
  const int n = n0 + (lane & 15);
  if (n >= N) return;  // columns computed from clamped B rows are dropped here
  const long mb = m0 + ((lane < 16) ? 0 : 8);
  const float bv = bias ? bias[n] : 0.f;
#pragma unroll
  for (int v = 0; v < 8; ++v) {
    float x = acc[v] + bv;
    if (act == 1) x = fmaxf(x, 0.f);
    else if (act == 2) x = tanhf(x);
    const long idx = (mb + v) * (long)N + n;
    if (C32) C32[idx] = x;
    if (C16) C16[idx] = (half_t)x;
  }
}

// ------------------------- generic WMMA GEMM -------------------------
// C[M,N] = act(A[M,K] @ W[N,K]^T + bias). One 32x32 tile per wave.
// Out-of-range N columns are CLAMPED (not masked) at the B rows: a WMMA
// D-column depends only on its own B-column, and clamped columns are never
// stored -- this keeps the K-loop free of exec-mask manipulation.
__global__ void k_gemm(const half_t* __restrict__ A, long lda,
                       const half_t* __restrict__ W, long ldw,
                       const float* __restrict__ bias,
                       float* __restrict__ C32, half_t* __restrict__ C16,
                       int M, int N, int K, int act) {
  const long wave = ((long)blockIdx.x * blockDim.x + threadIdx.x) >> 5;
  const int  nt   = (N + 31) >> 5;
  const long mt   = (long)M >> 5;
  if (wave >= mt * (long)nt) return;
  const long m0 = (wave / nt) * 32;
  const int  n0 = (int)(wave % nt) * 32;

  const int lane = threadIdx.x & 31;
  const int lrow = lane & 15;
  const int ka   = (lane < 16) ? 0 : 16;
  const int kb   = (lane < 16) ? 0 : 16;
  // hoisted per-lane base pointers (loop-invariant)
  const half_t* A0 = A + (m0 + lrow) * lda + ka;
  const half_t* A1 = A0 + 16 * lda;
  int nb0 = n0 + lrow;      if (nb0 > N - 1) nb0 = N - 1;
  int nb1 = n0 + 16 + lrow; if (nb1 > N - 1) nb1 = N - 1;
  const half_t* B0 = W + (long)nb0 * ldw + kb;
  const half_t* B1 = W + (long)nb1 * ldw + kb;

  v8f a00 = vzero8(), a01 = vzero8(), a10 = vzero8(), a11 = vzero8();
  for (int k = 0; k < K; k += 32) {
    if (k + 32 < K) {  // prefetch next K-chunk of the A strip (global_prefetch_b8)
      __builtin_prefetch(A0 + k + 32, 0, 1);
      __builtin_prefetch(A1 + k + 32, 0, 1);
    }
    v16h x0 = pack_a(A0 + k);
    v16h x1 = pack_a(A1 + k);
    v16h w0 = pack_b(B0 + k);
    v16h w1 = pack_b(B1 + k);
    a00 = wmma_f16(x0, w0, a00);
    a01 = wmma_f16(x0, w1, a01);
    a10 = wmma_f16(x1, w0, a10);
    a11 = wmma_f16(x1, w1, a11);
  }
  gemm_store(a00, C32, C16, bias, m0,      n0,      N, act);
  gemm_store(a01, C32, C16, bias, m0,      n0 + 16, N, act);
  gemm_store(a10, C32, C16, bias, m0 + 16, n0,      N, act);
  gemm_store(a11, C32, C16, bias, m0 + 16, n0 + 16, N, act);
}

// ------------------------- gathers / converts -------------------------
__global__ void k_cvt16(const float* __restrict__ s, half_t* __restrict__ d, long n) {
  long i = (long)blockIdx.x * blockDim.x + threadIdx.x;
  if (i < n) d[i] = (half_t)s[i];
}

__global__ void k_transpose_cvt(const float* __restrict__ s, half_t* __restrict__ d,
                                int R, int C) {
  long i = (long)blockIdx.x * blockDim.x + threadIdx.x;
  if (i >= (long)R * C) return;
  int r = (int)(i / C), c = (int)(i % C);
  d[(long)c * R + r] = (half_t)s[i];
}

__global__ void k_gather16(const int* __restrict__ tok, const float* __restrict__ emb,
                           half_t* __restrict__ out, long ntok) {
  long i = (long)blockIdx.x * blockDim.x + threadIdx.x;
  if (i >= (ntok << 7)) return;
  long t = i >> 7; int e = (int)(i & 127);
  out[i] = (half_t)emb[((long)tok[t] << 7) + e];
}

__global__ void k_gather_pe(const int* __restrict__ tok, const float* __restrict__ emb,
                            half_t* __restrict__ o16, float* __restrict__ o32, long ntok) {
  long i = (long)blockIdx.x * blockDim.x + threadIdx.x;
  if (i >= (ntok << 7)) return;
  long t = i >> 7; int e = (int)(i & 127);
  int s = (int)(t % mc::SC);
  float div = expf((float)(e >> 1) * (-logf(10000.f) / (float)mc::E));
  float ang = (float)s * div;
  float pe = (e & 1) ? cosf(ang) : sinf(ang);
  float x = emb[((long)tok[t] << 7) + e] + pe;
  o16[i] = (half_t)x; o32[i] = x;
}

// ------------------------- GRU gates (SFE) -------------------------
__global__ void k_gru_gates(const float* __restrict__ gi, const float* __restrict__ gh,
                            const float* __restrict__ bhh0,  // non-null => h_prev == 0
                            float* __restrict__ h32, half_t* __restrict__ h16,
                            float* __restrict__ runmax, int initmax, long rows) {
  long i = (long)blockIdx.x * blockDim.x + threadIdx.x;
  if (i >= rows * mc::E) return;
  long r = i >> 7; int e = (int)(i & 127);
  long b3 = r * (3L * mc::E);
  float ir = gi[b3 + e], iz = gi[b3 + mc::E + e], in_ = gi[b3 + 2 * mc::E + e];
  float hr, hz, hn_, hp;
  if (bhh0) { hr = bhh0[e]; hz = bhh0[mc::E + e]; hn_ = bhh0[2 * mc::E + e]; hp = 0.f; }
  else      { hr = gh[b3 + e]; hz = gh[b3 + mc::E + e]; hn_ = gh[b3 + 2 * mc::E + e]; hp = h32[i]; }
  float rg = sigmoidf(ir + hr);
  float zg = sigmoidf(iz + hz);
  float ng = tanhf(in_ + rg * hn_);
  float hv = (1.f - zg) * ng + zg * hp;
  h32[i] = hv; h16[i] = (half_t)hv;
  runmax[i] = initmax ? hv : fmaxf(runmax[i], hv);
}

__global__ void k_rowdot_tanh(const float* __restrict__ X, const float* __restrict__ w,
                              float* __restrict__ s, long rows, int dim) {
  long wv = ((long)blockIdx.x * blockDim.x + threadIdx.x) >> 5;
  int lane = threadIdx.x & 31;
  if (wv >= rows) return;
  float acc = 0.f;
  for (int j = lane; j < dim; j += 32) acc += X[wv * dim + j] * w[j];
  for (int o = 16; o; o >>= 1) acc += __shfl_xor(acc, o, 32);
  if (lane == 0) s[wv] = tanhf(acc);
}

__global__ void k_s0(const float* __restrict__ sent_b, const float* __restrict__ ctx_w,
                     float* __restrict__ s0) {
  if (threadIdx.x == 0 && blockIdx.x == 0) {
    float a = 0.f;
    for (int j = 0; j < mc::E; ++j) a += tanhf(sent_b[j]) * ctx_w[j];
    s0[0] = tanhf(a);
  }
}

__global__ void k_rootscale(const float* __restrict__ s, const float* __restrict__ s0,
                            float* __restrict__ h32, half_t* __restrict__ h16, long rows) {
  long i = (long)blockIdx.x * blockDim.x + threadIdx.x;
  if (i >= rows * mc::E) return;
  long r = i >> 7;
  float w = 1.f / (1.f + (float)(rows - 1) * expf(s0[0] - s[r]));
  float hv = w * h32[i];
  h32[i] = hv; h16[i] = (half_t)hv;
}

__global__ void k_maxns(const float* __restrict__ nm, float* __restrict__ out) {
  long i = (long)blockIdx.x * blockDim.x + threadIdx.x;
  if (i >= (long)mc::Bn * mc::E) return;
  long b = i >> 7; int e = (int)(i & 127);
  float m = 0.f;  // clamp-at-zero folded into the max
  for (int ns = 0; ns < mc::NS; ++ns)
    m = fmaxf(m, nm[((b * mc::NS + ns) << 7) + e]);
  out[i] = m;
}

// ------------------------- layernorm (+residual) -------------------------
__global__ void k_layernorm(const float* __restrict__ resid, const float* __restrict__ add,
                            const float* __restrict__ w, const float* __restrict__ b,
                            float* __restrict__ o32, half_t* __restrict__ o16, int dim) {
  __shared__ float red[128];
  const long r = blockIdx.x;
  const int tid = threadIdx.x;
  const int cnt = dim >> 7;  // 1 or 3
  float loc[3];
  float sum = 0.f;
  for (int j = 0; j < cnt; ++j) {
    long idx = r * dim + tid + (j << 7);
    float v = add[idx] + (resid ? resid[idx] : 0.f);
    loc[j] = v; sum += v;
  }
  red[tid] = sum; __syncthreads();
  for (int st = 64; st > 0; st >>= 1) { if (tid < st) red[tid] += red[tid + st]; __syncthreads(); }
  float mu = red[0] / (float)dim; __syncthreads();
  float vs = 0.f;
  for (int j = 0; j < cnt; ++j) { float d_ = loc[j] - mu; vs += d_ * d_; }
  red[tid] = vs; __syncthreads();
  for (int st = 64; st > 0; st >>= 1) { if (tid < st) red[tid] += red[tid + st]; __syncthreads(); }
  float inv = rsqrtf(red[0] / (float)dim + 1e-5f);
  for (int j = 0; j < cnt; ++j) {
    int col = tid + (j << 7);
    float y = (loc[j] - mu) * inv * w[col] + b[col];
    long idx = r * dim + col;
    if (o32) o32[idx] = y;
    if (o16) o16[idx] = (half_t)y;
  }
}

// ------------------------- attention (S=128), per (batch,head) WG -------------------------
__global__ void k_attn128(const half_t* __restrict__ qkv, half_t* __restrict__ o) {
  extern __shared__ char smem[];
  half_t* qs = (half_t*)smem;                        // [128][32]
  half_t* ks = qs + mc::SC * mc::DH;                 // [128][32]
  half_t* vT = ks + mc::SC * mc::DH;                 // [32][128]
  float*  Sc = (float*)(vT + mc::DH * mc::SC);       // [128][128] f32
  half_t* At = (half_t*)(Sc + mc::SC * mc::SC);      // [128][128] f16
  const int b = blockIdx.x >> 2;
  const int h = blockIdx.x & 3;
  const long base = (long)b * mc::SC;
  const int tid = threadIdx.x, wv = tid >> 5, lane = tid & 31;
  {
    const half_t* row = qkv + (base + tid) * (long)mc::E3 + h * mc::DH;
    for (int d = 0; d < mc::DH; ++d) {
      qs[tid * mc::DH + d] = row[d];
      ks[tid * mc::DH + d] = row[mc::E + d];
      vT[d * mc::SC + tid] = row[2 * mc::E + d];
    }
  }
  __syncthreads();
  const float scale = 0.17677669529663687f;  // 1/sqrt(32)
  for (int ti = wv; ti < 64; ti += 4) {
    int qt = ti >> 3, kt = ti & 7;
    v16h a = wmma_load_a(qs + qt * 16 * mc::DH, mc::DH);
    v16h w = wmma_load_bu(ks, mc::DH, kt * 16, 0);
    v8f c = vzero8();
    c = wmma_f16(a, w, c);
    int n = kt * 16 + (lane & 15), mb = qt * 16 + ((lane < 16) ? 0 : 8);
#pragma unroll
    for (int v = 0; v < 8; ++v) Sc[(mb + v) * mc::SC + n] = c[v] * scale;
  }
  __syncthreads();
  {
    float* row = Sc + tid * mc::SC;
    float mx = -3.0e38f;
    for (int j = 0; j < mc::SC; ++j) mx = fmaxf(mx, row[j]);
    float sum = 0.f;
    for (int j = 0; j < mc::SC; ++j) { float e = expf(row[j] - mx); row[j] = e; sum += e; }
    float inv = 1.f / sum;
    for (int j = 0; j < mc::SC; ++j) At[tid * mc::SC + j] = (half_t)(row[j] * inv);
  }
  __syncthreads();
  for (int ti = wv; ti < 16; ti += 4) {
    int rt = ti >> 1, ct = ti & 1;
    v8f c = vzero8();
    for (int k = 0; k < mc::SC; k += 32) {
      v16h a = wmma_load_a(At + rt * 16 * mc::SC + k, mc::SC);
      v16h w = wmma_load_bu(vT, mc::SC, ct * 16, k);
      c = wmma_f16(a, w, c);
    }
    int n = ct * 16 + (lane & 15), mb = rt * 16 + ((lane < 16) ? 0 : 8);
#pragma unroll
    for (int v = 0; v < 8; ++v)
      o[(base + mb + v) * (long)mc::E + h * mc::DH + n] = (half_t)c[v];
  }
}

// ------------------------- tiny attention (S=3), one wave per (b,h) -------------------------
__global__ void k_attn3(const half_t* __restrict__ qkv, half_t* __restrict__ o) {
  const int gw = (int)(((long)blockIdx.x * blockDim.x + threadIdx.x) >> 5);
  const int lane = threadIdx.x & 31;
  if (gw >= mc::Bn * mc::Hh) return;
  const int b = gw >> 2, h = gw & 3;
  float q[3], kk[3], vv[3];
  for (int s = 0; s < 3; ++s) {
    const half_t* row = qkv + (long)(b * 3 + s) * mc::E3 + h * mc::DH;
    q[s]  = (float)row[lane];
    kk[s] = (float)row[mc::E + lane];
    vv[s] = (float)row[2 * mc::E + lane];
  }
  float sc[3][3];
  const float scale = 0.17677669529663687f;
  for (int i = 0; i < 3; ++i)
    for (int j = 0; j < 3; ++j) {
      float p = q[i] * kk[j];
      for (int off = 16; off; off >>= 1) p += __shfl_xor(p, off, 32);
      sc[i][j] = p * scale;
    }
  for (int i = 0; i < 3; ++i) {
    float mx = fmaxf(sc[i][0], fmaxf(sc[i][1], sc[i][2]));
    float e0 = expf(sc[i][0] - mx), e1 = expf(sc[i][1] - mx), e2 = expf(sc[i][2] - mx);
    float inv = 1.f / (e0 + e1 + e2);
    float ov = (e0 * vv[0] + e1 * vv[1] + e2 * vv[2]) * inv;
    o[(long)(b * 3 + i) * mc::E + h * mc::DH + lane] = (half_t)ov;
  }
}

// ------------------------- fused FFN: relu(x@W1.T+b1)@W2.T+b2 -------------------------
__global__ void k_ffn(const half_t* __restrict__ X, const half_t* __restrict__ W1,
                      const float* __restrict__ b1, const half_t* __restrict__ W2,
                      const float* __restrict__ b2, float* __restrict__ out) {
  extern __shared__ char smem[];
  half_t* xs  = (half_t*)smem;          // [16][E]
  half_t* mid = xs + 16 * mc::E;        // [16][FF] (64KB) in LDS, never touches HBM
  const long r0 = (long)blockIdx.x * 16;
  const int tid = threadIdx.x, wv = tid >> 5, lane = tid & 31;
  for (int i = tid; i < 16 * mc::E; i += 256)
    xs[i] = X[(r0 + (i >> 7)) * mc::E + (i & 127)];
  __syncthreads();
  for (int ct = wv; ct < mc::FF / 16; ct += 8) {
    v8f c = vzero8();
    for (int k = 0; k < mc::E; k += 32) {
      v16h a = wmma_load_a(xs + k, mc::E);
      v16h w = wmma_load_bu(W1, mc::E, ct * 16, k);
      c = wmma_f16(a, w, c);
    }
    int n = ct * 16 + (lane & 15);
    int mb = (lane < 16) ? 0 : 8;
    float bv = b1[n];
#pragma unroll
    for (int v = 0; v < 8; ++v)
      mid[(mb + v) * mc::FF + n] = (half_t)fmaxf(c[v] + bv, 0.f);
  }
  __syncthreads();
  {
    int ct = wv;  // 8 waves, 8 output col tiles
    v8f c = vzero8();
    for (int k = 0; k < mc::FF; k += 32) {
      v16h a = wmma_load_a(mid + k, mc::FF);
      v16h w = wmma_load_bu(W2, mc::FF, ct * 16, k);
      c = wmma_f16(a, w, c);
    }
    int n = ct * 16 + (lane & 15);
    int mb = (lane < 16) ? 0 : 8;
    float bv = b2[n];
#pragma unroll
    for (int v = 0; v < 8; ++v) out[(r0 + mb + v) * mc::E + n] = c[v] + bv;
  }
}

// ------------------------- persistent DFFE GRU: 512 serial steps, one WG -------------------------
__global__ void __launch_bounds__(1024) k_dffe_gru(
    const half_t* __restrict__ X,    // [B, TD, E] f16 (lda = TD*E)
    const half_t* __restrict__ Wih,  // [3E, E]
    const half_t* __restrict__ Whh,  // [3E, E]
    const float* __restrict__ bih, const float* __restrict__ bhh,
    half_t* __restrict__ mean16) {   // [B, E]
  __shared__ half_t hsh[mc::E * mc::E];  // 32KB hidden-state ping buffer
  const int tid = threadIdx.x, wv = tid >> 5, lane = tid & 31;
  for (int i = tid; i < mc::E * mc::E; i += 1024) hsh[i] = (half_t)0.f;

  v8f hreg[2], macc[2];
  hreg[0] = vzero8(); hreg[1] = vzero8();
  macc[0] = vzero8(); macc[1] = vzero8();
  float bR[2], bZ[2], bIN[2], bHN[2];
  int rt[2], ct[2];
  for (int j = 0; j < 2; ++j) {
    int tile = wv * 2 + j;            // 64 tiles of the [128,128] h-state grid
    rt[j] = tile >> 3; ct[j] = tile & 7;
    int n = ct[j] * 16 + (lane & 15);
    bR[j]  = bih[n] + bhh[n];
    bZ[j]  = bih[mc::E + n] + bhh[mc::E + n];
    bIN[j] = bih[2 * mc::E + n];
    bHN[j] = bhh[2 * mc::E + n];
  }
  __syncthreads();
  const long ldx = (long)mc::TD * mc::E;
  for (int t = 0; t < mc::TD; ++t) {
    for (int j = 0; j < 2; ++j) {
      v8f gR = vzero8(), gZ = vzero8(), gN = vzero8();
      v8f hR = vzero8(), hZ = vzero8(), hN = vzero8();
      const half_t* Ab = X + (long)(rt[j] * 16) * ldx + (long)t * mc::E;
      const half_t* Hb = hsh + rt[j] * 16 * mc::E;
      for (int k = 0; k < mc::E; k += 32) {
        v16h ax = wmma_load_a(Ab + k, ldx);
        v16h ah = wmma_load_a(Hb + k, mc::E);
        v16h w0 = wmma_load_bu(Wih, mc::E, ct[j] * 16, k);
        v16h w1 = wmma_load_bu(Wih, mc::E, mc::E + ct[j] * 16, k);
        v16h w2 = wmma_load_bu(Wih, mc::E, 2 * mc::E + ct[j] * 16, k);
        gR = wmma_f16(ax, w0, gR);
        gZ = wmma_f16(ax, w1, gZ);
        gN = wmma_f16(ax, w2, gN);
        v16h u0 = wmma_load_bu(Whh, mc::E, ct[j] * 16, k);
        v16h u1 = wmma_load_bu(Whh, mc::E, mc::E + ct[j] * 16, k);
        v16h u2 = wmma_load_bu(Whh, mc::E, 2 * mc::E + ct[j] * 16, k);
        hR = wmma_f16(ah, u0, hR);
        hZ = wmma_f16(ah, u1, hZ);
        hN = wmma_f16(ah, u2, hN);
      }
      v8f hn;
#pragma unroll
      for (int v = 0; v < 8; ++v) {
        float rg = sigmoidf(gR[v] + hR[v] + bR[j]);
        float zg = sigmoidf(gZ[v] + hZ[v] + bZ[j]);
        float ng = tanhf(gN[v] + bIN[j] + rg * (hN[v] + bHN[j]));
        hn[v] = (1.f - zg) * ng + zg * hreg[j][v];  // hreg == this tile of h_prev
      }
      hreg[j] = hn;
      macc[j] = macc[j] + hn;
    }
    __syncthreads();  // all hsh reads of step t done
    for (int j = 0; j < 2; ++j) {
      int n = ct[j] * 16 + (lane & 15);
      int mb = rt[j] * 16 + ((lane < 16) ? 0 : 8);
#pragma unroll
      for (int v = 0; v < 8; ++v) hsh[(mb + v) * mc::E + n] = (half_t)hreg[j][v];
    }
    __syncthreads();  // hsh updated for step t+1
  }
  const float inv = 1.f / (float)mc::TD;
  for (int j = 0; j < 2; ++j) {
    int n = ct[j] * 16 + (lane & 15);
    int mb = rt[j] * 16 + ((lane < 16) ? 0 : 8);
#pragma unroll
    for (int v = 0; v < 8; ++v) mean16[(mb + v) * mc::E + n] = (half_t)(macc[j][v] * inv);
  }
}

// ------------------------- small reductions / glue -------------------------
__global__ void k_mean_seq(const float* __restrict__ x, float* __restrict__ m32,
                           half_t* __restrict__ m16) {
  long i = (long)blockIdx.x * blockDim.x + threadIdx.x;
  if (i >= mc::PB * mc::E) return;
  long p = i >> 7; int e = (int)(i & 127);
  float s = 0.f;
  for (int t = 0; t < mc::SC; ++t) s += x[((p * mc::SC + t) << 7) + e];
  float v = s / (float)mc::SC;
  m32[i] = v; m16[i] = (half_t)v;
}

__global__ void k_stack3(const float* __restrict__ a, const float* __restrict__ b_,
                         const float* __restrict__ c, float* __restrict__ o32,
                         half_t* __restrict__ o16) {
  long i = (long)blockIdx.x * blockDim.x + threadIdx.x;
  if (i >= (long)mc::Bn * mc::E) return;
  long b = i >> 7; int e = (int)(i & 127);
  long base = b * 3 * mc::E;
  float v0 = a[i], v1 = b_[i], v2 = c[i];
  o32[base + e] = v0;             o16[base + e] = (half_t)v0;
  o32[base + mc::E + e] = v1;     o16[base + mc::E + e] = (half_t)v1;
  o32[base + 2 * mc::E + e] = v2; o16[base + 2 * mc::E + e] = (half_t)v2;
}

// ============================ host orchestration ============================
extern "C" void kernel_launch(void* const* d_in, const int* in_sizes, int n_in,
                              void* d_out, int out_size, void* d_ws, size_t ws_size,
                              hipStream_t stream) {
  using namespace mc;
  const int*   subtree   = (const int*)d_in[0];
  const int*   cfgp      = (const int*)d_in[1];
  const int*   dfg       = (const int*)d_in[2];
  const float* sfe_emb   = (const float*)d_in[3];
  const float* sfe_wih   = (const float*)d_in[4];
  const float* sfe_whh   = (const float*)d_in[5];
  const float* sfe_bih   = (const float*)d_in[6];
  const float* sfe_bhh   = (const float*)d_in[7];
  const float* sent_w    = (const float*)d_in[8];
  const float* sent_b    = (const float*)d_in[9];
  const float* ctx_w     = (const float*)d_in[10];
  const float* cfpe_emb  = (const float*)d_in[11];
  const float* cfpe_qkv_w = (const float*)d_in[12];
  const float* cfpe_qkv_b = (const float*)d_in[13];
  const float* cfpe_out_w = (const float*)d_in[14];
  const float* cfpe_out_b = (const float*)d_in[15];
  const float* cfpe_ln1_w = (const float*)d_in[16];
  const float* cfpe_ln1_b = (const float*)d_in[17];
  const float* cfpe_ln2_w = (const float*)d_in[18];
  const float* cfpe_ln2_b = (const float*)d_in[19];
  const float* cfpe_ff1_w = (const float*)d_in[20];
  const float* cfpe_ff1_b = (const float*)d_in[21];
  const float* cfpe_ff2_w = (const float*)d_in[22];
  const float* cfpe_ff2_b = (const float*)d_in[23];
  const float* glob_qkv_w = (const float*)d_in[24];
  const float* glob_qkv_b = (const float*)d_in[25];
  const float* glob_out_w = (const float*)d_in[26];
  const float* glob_out_b = (const float*)d_in[27];
  const float* glob_ln1_w = (const float*)d_in[28];
  const float* glob_ln1_b = (const float*)d_in[29];
  const float* glob_ln2_w = (const float*)d_in[30];
  const float* glob_ln2_b = (const float*)d_in[31];
  const float* glob_ff1_w = (const float*)d_in[32];
  const float* glob_ff1_b = (const float*)d_in[33];
  const float* glob_ff2_w = (const float*)d_in[34];
  const float* glob_ff2_b = (const float*)d_in[35];
  const float* dffe_emb   = (const float*)d_in[36];
  const float* dffe_wih   = (const float*)d_in[37];
  const float* dffe_whh   = (const float*)d_in[38];
  const float* dffe_bih   = (const float*)d_in[39];
  const float* dffe_bhh   = (const float*)d_in[40];
  const float* cfgfuse_w  = (const float*)d_in[41];
  const float* cfgfuse_b  = (const float*)d_in[42];
  const float* dfg_w      = (const float*)d_in[43];
  const float* dfg_b      = (const float*)d_in[44];
  const float* cls_ln_w   = (const float*)d_in[45];
  const float* cls_ln_b   = (const float*)d_in[46];
  const float* cls_l1_w   = (const float*)d_in[47];
  const float* cls_l1_b   = (const float*)d_in[48];
  const float* cls_l2_w   = (const float*)d_in[49];
  const float* cls_l2_b   = (const float*)d_in[50];
  (void)in_sizes; (void)n_in; (void)out_size; (void)ws_size;

  char* wsp = (char*)d_ws;
  size_t off = 0;
  auto alloc  = [&](size_t bytes) -> void* {
    void* p = wsp + off;
    off += (bytes + 255) & ~(size_t)255;
    return p;
  };
  auto allocH = [&](long n) { return (half_t*)alloc((size_t)n * sizeof(half_t)); };
  auto allocF = [&](long n) { return (float*)alloc((size_t)n * sizeof(float)); };
  auto grid1  = [&](long n) { return dim3((unsigned)((n + 255) / 256)); };

  auto cvt = [&](const float* src, long n) {
    half_t* dst = allocH(n);
    k_cvt16<<<grid1(n), dim3(256), 0, stream>>>(src, dst, n);
    return dst;
  };

  // ---- weights -> f16 (one-time per call, ~5.4MB total, L2-resident) ----
  half_t* w_sfe_wih = cvt(sfe_wih, 3L * E * E);
  half_t* w_sfe_whh = cvt(sfe_whh, 3L * E * E);
  half_t* w_sentT   = allocH((long)E * E);
  k_transpose_cvt<<<grid1((long)E * E), dim3(256), 0, stream>>>(sent_w, w_sentT, E, E);
  half_t* w_cq  = cvt(cfpe_qkv_w, 2L * 3 * E * E);
  half_t* w_co  = cvt(cfpe_out_w, 2L * E * E);
  half_t* w_cf1 = cvt(cfpe_ff1_w, 2L * FF * E);
  half_t* w_cf2 = cvt(cfpe_ff2_w, 2L * E * FF);
  half_t* w_gq  = cvt(glob_qkv_w, 2L * 3 * E * E);
  half_t* w_go  = cvt(glob_out_w, 2L * E * E);
  half_t* w_gf1 = cvt(glob_ff1_w, 2L * FF * E);
  half_t* w_gf2 = cvt(glob_ff2_w, 2L * E * FF);
  half_t* w_dih = cvt(dffe_wih, 3L * E * E);
  half_t* w_dhh = cvt(dffe_whh, 3L * E * E);
  half_t* w_fuse = cvt(cfgfuse_w, (long)E * 3 * E);
  half_t* w_dfg  = cvt(dfg_w, (long)E * E);
  half_t* w_l1   = cvt(cls_l1_w, (long)E * 3 * E);
  half_t* w_l2   = cvt(cls_l2_w, (long)Lc * E);

  auto gemm = [&](const half_t* A, long lda, const half_t* W, long ldw,
                  const float* bias, float* C32, half_t* C16,
                  long Mr, int N, int K, int act) {
    int nt = (N + 31) / 32;
    long waves = (Mr / 32) * nt;
    unsigned grid = (unsigned)((waves * 32 + 255) / 256);
    k_gemm<<<dim3(grid), dim3(256), 0, stream>>>(A, lda, W, ldw, bias, C32, C16,
                                                 (int)Mr, N, K, act);
  };
  auto ln = [&](const float* resid, const float* add, const float* w, const float* b,
                float* o32, half_t* o16, long rows, int dim) {
    k_layernorm<<<dim3((unsigned)rows), dim3(128), 0, stream>>>(resid, add, w, b, o32, o16, dim);
  };

  // ===================== SFE: recursive tree GRU =====================
  half_t* x16 = allocH(NSB * DEP * E);
  k_gather16<<<grid1(NSB * DEP * E), dim3(256), 0, stream>>>(subtree, sfe_emb, x16, NSB * DEP);
  float*  gi = allocF(NSB * 3 * E);
  float*  gh = allocF(NSB * 3 * E);
  float*  h32 = allocF(NSB * E);
  half_t* h16 = allocH(NSB * E);
  float*  runmax = allocF(NSB * E);
  float*  hsct = allocF(NSB * E);
  float*  svec = allocF(NSB);
  float*  s0   = allocF(64);
  const long NE = NSB * E;

  // leaf (depth DEP-1, h==0: gh collapses to bhh broadcast)
  gemm(x16 + (DEP - 1) * E, (long)DEP * E, w_sfe_wih, E, sfe_bih, gi, nullptr, NSB, 3 * E, E, 0);
  k_gru_gates<<<grid1(NE), dim3(256), 0, stream>>>(gi, nullptr, sfe_bhh, h32, h16, runmax, 1, NSB);
  // bottom-up mids: depth DEP-2 .. 1
  for (int d = DEP - 2; d >= 1; --d) {
    gemm(x16 + d * E, (long)DEP * E, w_sfe_wih, E, sfe_bih, gi, nullptr, NSB, 3 * E, E, 0);
    gemm(h16, E, w_sfe_whh, E, sfe_bhh, gh, nullptr, NSB, 3 * E, E, 0);
    k_gru_gates<<<grid1(NE), dim3(256), 0, stream>>>(gi, gh, nullptr, h32, h16, runmax, 0, NSB);
  }
  // diagonal-softmax root weight
  gemm(h16, E, w_sentT, E, sent_b, hsct, nullptr, NSB, E, E, 2 /*tanh*/);
  k_rowdot_tanh<<<grid1(NSB * 32), dim3(256), 0, stream>>>(hsct, ctx_w, svec, NSB, E);
  k_s0<<<dim3(1), dim3(1), 0, stream>>>(sent_b, ctx_w, s0);
  k_rootscale<<<grid1(NE), dim3(256), 0, stream>>>(svec, s0, h32, h16, NSB);
  // root GRU (depth 0)
  gemm(x16, (long)DEP * E, w_sfe_wih, E, sfe_bih, gi, nullptr, NSB, 3 * E, E, 0);
  gemm(h16, E, w_sfe_whh, E, sfe_bhh, gh, nullptr, NSB, 3 * E, E, 0);
  k_gru_gates<<<grid1(NE), dim3(256), 0, stream>>>(gi, gh, nullptr, h32, h16, runmax, 0, NSB);
  float* h_sub = allocF((long)Bn * E);
  k_maxns<<<grid1((long)Bn * E), dim3(256), 0, stream>>>(runmax, h_sub);

  // ===================== CFPE: 2-layer transformer =====================
  half_t* cx16 = allocH(PT * E);
  float*  cx32 = allocF(PT * E);
  k_gather_pe<<<grid1(PT * E), dim3(256), 0, stream>>>(cfgp, cfpe_emb, cx16, cx32, PT);
  half_t* qkv16 = allocH(PT * 3 * E);
  half_t* o16   = allocH(PT * E);
  float*  t32   = allocF(PT * E);
  for (int l = 0; l < 2; ++l) {
    gemm(cx16, E, w_cq + (long)l * 3 * E * E, E, cfpe_qkv_b + l * 3 * E, nullptr, qkv16, PT, 3 * E, E, 0);
    k_attn128<<<dim3((unsigned)(PB * Hh)), dim3(128), 122880, stream>>>(qkv16, o16);
    gemm(o16, E, w_co + (long)l * E * E, E, cfpe_out_b + l * E, t32, nullptr, PT, E, E, 0);
    ln(cx32, t32, cfpe_ln1_w + l * E, cfpe_ln1_b + l * E, cx32, cx16, PT, E);
    k_ffn<<<dim3((unsigned)(PT / 16)), dim3(256), 69632, stream>>>(
        cx16, w_cf1 + (long)l * FF * E, cfpe_ff1_b + l * FF,
        w_cf2 + (long)l * E * FF, cfpe_ff2_b + l * E, t32);
    ln(cx32, t32, cfpe_ln2_w + l * E, cfpe_ln2_b + l * E, cx32, cx16, PT, E);
  }
  float*  mean32 = allocF(PB * E);
  half_t* mean16 = allocH(PB * E);
  k_mean_seq<<<grid1(PB * E), dim3(256), 0, stream>>>(cx32, mean32, mean16);
  float* h_cfg = allocF((long)Bn * E);
  gemm(mean16, 3 * E, w_fuse, 3 * E, cfgfuse_b, h_cfg, nullptr, Bn, E, 3 * E, 0);

  // ===================== DFFE: persistent serial GRU =====================
  half_t* dx16 = allocH((long)Bn * TD * E);
  k_gather16<<<grid1((long)Bn * TD * E), dim3(256), 0, stream>>>(dfg, dffe_emb, dx16, (long)Bn * TD);
  half_t* dmean16 = allocH((long)Bn * E);
  k_dffe_gru<<<dim3(1), dim3(1024), 0, stream>>>(dx16, w_dih, w_dhh, dffe_bih, dffe_bhh, dmean16);
  float* h_dfg = allocF((long)Bn * E);
  gemm(dmean16, E, w_dfg, E, dfg_b, h_dfg, nullptr, Bn, E, E, 0);

  // ===================== global fusion + classifier =====================
  float*  hc32 = allocF(3L * Bn * E);
  half_t* hc16 = allocH(3L * Bn * E);
  k_stack3<<<grid1((long)Bn * E), dim3(256), 0, stream>>>(h_sub, h_cfg, h_dfg, hc32, hc16);
  half_t* gqkv = allocH(3L * Bn * 3 * E);
  half_t* go   = allocH(3L * Bn * E);
  float*  gt32 = allocF(3L * Bn * E);
  for (int l = 0; l < 2; ++l) {
    gemm(hc16, E, w_gq + (long)l * 3 * E * E, E, glob_qkv_b + l * 3 * E, nullptr, gqkv, 3L * Bn, 3 * E, E, 0);
    k_attn3<<<dim3((unsigned)(Bn * Hh / 4)), dim3(128), 0, stream>>>(gqkv, go);
    gemm(go, E, w_go + (long)l * E * E, E, glob_out_b + l * E, gt32, nullptr, 3L * Bn, E, E, 0);
    ln(hc32, gt32, glob_ln1_w + l * E, glob_ln1_b + l * E, hc32, hc16, 3L * Bn, E);
    k_ffn<<<dim3((unsigned)(3 * Bn / 16)), dim3(256), 69632, stream>>>(
        hc16, w_gf1 + (long)l * FF * E, glob_ff1_b + l * FF,
        w_gf2 + (long)l * E * FF, glob_ff2_b + l * E, gt32);
    ln(hc32, gt32, glob_ln2_w + l * E, glob_ln2_b + l * E, hc32, hc16, 3L * Bn, E);
  }
  half_t* y16 = allocH((long)Bn * 3 * E);
  ln(nullptr, hc32, cls_ln_w, cls_ln_b, nullptr, y16, Bn, 3 * E);  // [B, 3E] contiguous view
  half_t* y1 = allocH((long)Bn * E);
  gemm(y16, 3 * E, w_l1, 3 * E, cls_l1_b, nullptr, y1, Bn, E, 3 * E, 1 /*relu*/);
  gemm(y1, E, w_l2, E, cls_l2_b, (float*)d_out, nullptr, Bn, Lc, E, 0);  // [B, 104]
}
